// DualViewGNN_84593675862506
// MI455X (gfx1250) — compile-verified, matrix operand order checked by reading, not run on hardware
//
#include <hip/hip_runtime.h>
#include <math.h>
#include <stdint.h>

// Problem constants (match reference)
#define DD   768
#define RR   8
#define NN   8192
#define EE   32768
#define ECC  16384
#define PP   512
#define EDD  4096

// B-tile LDS staging: chunk of K staged per block, double buffered.
#define KC   256
#define BPAD 8     // pad row to 264 shorts = 132 dwords -> conflict-free frag reads

typedef __bf16 bf16_t;
typedef bf16_t v16bf __attribute__((ext_vector_type(16)));
typedef float  v8f   __attribute__((ext_vector_type(8)));

// ---------- small device helpers ----------
__device__ __forceinline__ float leakyf(float x) { return x >= 0.f ? x : 0.2f * x; }
__device__ __forceinline__ float eluf(float x)   { return x > 0.f ? x : (expf(x) - 1.f); }
__device__ __forceinline__ unsigned short f2bf(float f) {
  unsigned int u = __float_as_uint(f);
  u += 0x7fffu + ((u >> 16) & 1u);          // round-to-nearest-even
  return (unsigned short)(u >> 16);
}
__device__ __forceinline__ float bf2f(unsigned short h) {
  return __uint_as_float(((unsigned int)h) << 16);
}
// monotonic float<->uint mapping so atomicMax(uint) == float max
__device__ __forceinline__ unsigned int fkey(float f) {
  unsigned int u = __float_as_uint(f);
  return (u & 0x80000000u) ? ~u : (u | 0x80000000u);
}
__device__ __forceinline__ float fkeyinv(unsigned int k) {
  unsigned int u = (k & 0x80000000u) ? (k & 0x7fffffffu) : ~k;
  return __uint_as_float(u);
}

// Load a 16-element bf16 WMMA fragment (A or B operand) from a row pointer
// (global or LDS). ISA 16-bit operand layout:
//   lane<16 -> K = k0 + {0..7, 16..23}, lane>=16 -> K = k0 + 8 + {0..7, 16..23}.
__device__ __forceinline__ v16bf load_frag(const unsigned short* p, int kbase) {
  union { unsigned int u[8]; v16bf v; } r;
#pragma unroll
  for (int j = 0; j < 4; ++j) r.u[j]     = *(const unsigned int*)(p + kbase + 2 * j);
#pragma unroll
  for (int j = 0; j < 4; ++j) r.u[4 + j] = *(const unsigned int*)(p + kbase + 16 + 2 * j);
  return r.v;
}

// epilogue for one 16x16 accumulator tile
__device__ __forceinline__ void gemm_epilogue(
    const v8f& acc, int tileM, int ncol, int laneHi, int Nc, int act,
    const float* __restrict__ bias, float* __restrict__ outF,
    unsigned short* __restrict__ outB, const float* __restrict__ rowscale,
    const int* __restrict__ scatter, float* __restrict__ accum) {
  const float bv = bias ? bias[ncol] : 0.f;
#pragma unroll
  for (int v = 0; v < 8; ++v) {
    const int r = tileM + v + laneHi * 8;   // C/D layout: vgpr v -> row v (+8 for hi lanes)
    float val = acc[v] + bv;
    if (act == 1) val = leakyf(val);
    else if (act == 2) val = fmaxf(val, 0.f);
    if (outF) outF[(size_t)r * Nc + ncol] = val;
    if (outB) outB[(size_t)r * Nc + ncol] = f2bf(val);
    if (accum) {
      const float s = rowscale[r];
      if (s != 0.f) atomicAdd(&accum[(size_t)scatter[r] * Nc + ncol], s * val);
    }
  }
}

// ---------- WMMA gather-GEMM (the workhorse) ----------
// C[M,Nc] = act( gatherA[M,K] @ Wt^T + bias ), Wt stored transposed [Nc,K] bf16.
// A rows gathered via idx0 (and idx1 for the concat half when A1 != null, split
// at Ksplit). Each wave register-blocks a 32x32 output (2x2 subtiles): 4
// independent WMMA chains hide the bf16-WMMA hazard window. All 8 waves of a
// block share the same 32 B-columns, so the B tile is staged through LDS in
// double-buffered KC-wide chunks (global loads of chunk c+1 overlap WMMAs on
// chunk c), removing the 8x redundant B fetch. Requires M % 256 == 0,
// K % KC == 0, Ksplit % KC == 0.
// Output modes: dense f32 store, dense bf16 store, and/or attention-weighted
// row scatter-add: accum[scatter[m]] += rowscale[m] * C[m].
__global__ void __launch_bounds__(256)
k_wmma_gemm(const unsigned short* __restrict__ A0, const unsigned short* __restrict__ A1,
            const int* __restrict__ idx0, const int* __restrict__ idx1, int lda,
            const unsigned short* __restrict__ Wt, const float* __restrict__ bias,
            int M, int Nc, int K, int Ksplit, int act,
            float* __restrict__ outF, unsigned short* __restrict__ outB,
            const float* __restrict__ rowscale, const int* __restrict__ scatter,
            float* __restrict__ accum) {
  __shared__ alignas(16) unsigned short smB[2][32][KC + BPAD];
  const int lane = threadIdx.x & 31;
  const int wave = threadIdx.x >> 5;
  const int tileM = (blockIdx.y * 8 + wave) * 32;
  const int tileN = blockIdx.x * 32;
  const int laneHi = lane >> 4;           // 0 or 1
  const int kbase  = laneHi * 8;
  const int m0 = tileM + (lane & 15);
  const int m1 = m0 + 16;
  const int n0 = tileN + (lane & 15);
  const int n1 = n0 + 16;

  const unsigned short* rA0m0 = A0 + (size_t)(idx0 ? idx0[m0] : m0) * lda;
  const unsigned short* rA0m1 = A0 + (size_t)(idx0 ? idx0[m1] : m1) * lda;
  const unsigned short* rA1m0 =
      A1 ? (A1 + (size_t)(idx1 ? idx1[m0] : m0) * lda) : nullptr;
  const unsigned short* rA1m1 =
      A1 ? (A1 + (size_t)(idx1 ? idx1[m1] : m1) * lda) : nullptr;

  const int u4PerRow = KC / 8;            // uint4 per staged row
  const int totU4    = 32 * u4PerRow;

  // preload B chunk 0 into buffer 0
  for (int idx = threadIdx.x; idx < totU4; idx += 256) {
    const int row = idx / u4PerRow, c16 = idx % u4PerRow;
    *(uint4*)&smB[0][row][c16 * 8] =
        *(const uint4*)(Wt + (size_t)(tileN + row) * K + c16 * 8);
  }
  __syncthreads();

  v8f acc00 = {0.f,0.f,0.f,0.f,0.f,0.f,0.f,0.f};
  v8f acc01 = acc00, acc10 = acc00, acc11 = acc00;

  const int nchunks = K / KC;
  for (int c = 0; c < nchunks; ++c) {
    const int buf = c & 1;
    if (c + 1 < nchunks) {                // prefetch next B chunk into other buffer
      const int k0n = (c + 1) * KC;
      for (int idx = threadIdx.x; idx < totU4; idx += 256) {
        const int row = idx / u4PerRow, c16 = idx % u4PerRow;
        *(uint4*)&smB[buf ^ 1][row][c16 * 8] =
            *(const uint4*)(Wt + (size_t)(tileN + row) * K + k0n + c16 * 8);
      }
    }
    const int k0c = c * KC;
#pragma unroll
    for (int kk = 0; kk < KC; kk += 32) {
      const int k0 = k0c + kk;
      const unsigned short *ra0, *ra1;
      if (k0 < Ksplit) { ra0 = rA0m0 + k0; ra1 = rA0m1 + k0; }
      else             { ra0 = rA1m0 + (k0 - Ksplit); ra1 = rA1m1 + (k0 - Ksplit); }
      v16bf a0 = load_frag(ra0, kbase);
      v16bf a1 = load_frag(ra1, kbase);
      v16bf b0 = load_frag(&smB[buf][lane & 15][kk], kbase);
      v16bf b1 = load_frag(&smB[buf][(lane & 15) + 16][kk], kbase);
      acc00 = __builtin_amdgcn_wmma_f32_16x16x32_bf16(false, a0, false, b0, (short)0, acc00, false, false);
      acc01 = __builtin_amdgcn_wmma_f32_16x16x32_bf16(false, a0, false, b1, (short)0, acc01, false, false);
      acc10 = __builtin_amdgcn_wmma_f32_16x16x32_bf16(false, a1, false, b0, (short)0, acc10, false, false);
      acc11 = __builtin_amdgcn_wmma_f32_16x16x32_bf16(false, a1, false, b1, (short)0, acc11, false, false);
    }
    __syncthreads();                      // chunk consumed + next chunk staged
  }

  gemm_epilogue(acc00, tileM,      n0, laneHi, Nc, act, bias, outF, outB, rowscale, scatter, accum);
  gemm_epilogue(acc01, tileM,      n1, laneHi, Nc, act, bias, outF, outB, rowscale, scatter, accum);
  gemm_epilogue(acc10, tileM + 16, n0, laneHi, Nc, act, bias, outF, outB, rowscale, scatter, accum);
  gemm_epilogue(acc11, tileM + 16, n1, laneHi, Nc, act, bias, outF, outB, rowscale, scatter, accum);
}

// ---------- utility kernels ----------
__global__ void k_zero(unsigned int* p, long long n) {
  long long i = (long long)blockIdx.x * 256 + threadIdx.x;
  if (i < n) p[i] = 0u;
}
__global__ void k_cvt(unsigned short* d, const float* s, long long n) {
  long long i = (long long)blockIdx.x * 256 + threadIdx.x;
  if (i < n) d[i] = f2bf(s[i]);
}
// dst[Nc][K] (bf16) = transpose of src[K][Nc] (f32)
__global__ void k_transpose_cvt(unsigned short* d, const float* s, int K, int Nc) {
  long long i = (long long)blockIdx.x * 256 + threadIdx.x;
  if (i >= (long long)K * Nc) return;
  int n = (int)(i / K), k = (int)(i % K);
  d[i] = f2bf(s[(size_t)k * Nc + n]);
}
// u[d] = WQ[d,:]@a1 + WK[d,:]@a2 ; u[D] = bQ@a1 + bK@a2 + a_b  (score collapse)
__global__ void k_compute_u(float* u, const float* WQ, const float* WK,
                            const float* bQ, const float* bK,
                            const float* aw, const float* ab, int D_) {
  int d = blockIdx.x * 256 + threadIdx.x;
  if (d < D_) {
    float s = 0.f;
    for (int j = 0; j < D_; ++j)
      s += WQ[(size_t)d * D_ + j] * aw[j] + WK[(size_t)d * D_ + j] * aw[D_ + j];
    u[d] = s;
  } else if (d == D_) {
    float s = ab[0];
    for (int j = 0; j < D_; ++j) s += bQ[j] * aw[j] + bK[j] * aw[D_ + j];
    u[D_] = s;
  }
}
__global__ void k_edge_score(float* e, const unsigned short* relb, const float* u,
                             int ne, int D_) {
  int i = blockIdx.x * 256 + threadIdx.x;
  if (i >= ne) return;
  const unsigned short* row = relb + (size_t)i * D_;
  float s = u[D_];
  for (int d = 0; d < D_; ++d) s += bf2f(row[d]) * u[d];
  e[i] = leakyf(s);
}
// segment softmax: 3 atomic passes
__global__ void k_seg_max(const float* e, const int* etype, int r, const int* seg,
                          unsigned int* key, int ne) {
  int i = blockIdx.x * 256 + threadIdx.x;
  if (i >= ne) return;
  if (etype && etype[i] != r) return;
  atomicMax(&key[seg[i]], fkey(e[i]));
}
__global__ void k_seg_expsum(const float* e, const int* etype, int r, const int* seg,
                             const unsigned int* key, float* sum, float* ex, int ne) {
  int i = blockIdx.x * 256 + threadIdx.x;
  if (i >= ne) return;
  if (etype && etype[i] != r) { ex[i] = 0.f; return; }
  float m = fkeyinv(key[seg[i]]);
  float x = expf(e[i] - m);
  ex[i] = x;
  atomicAdd(&sum[seg[i]], x);
}
__global__ void k_seg_norm(float* ex, const int* seg, const float* sum, int ne) {
  int i = blockIdx.x * 256 + threadIdx.x;
  if (i >= ne) return;
  float s = sum[seg[i]];
  ex[i] = (s > 0.f) ? ex[i] / s : 0.f;
}
// out = layernorm(x + f(h)) rowwise; mode 1: f=elu, mode 2: f=identity
__global__ void __launch_bounds__(256)
k_rowops_ln(const float* x, const float* h, int mode, const float* g, const float* b,
            float* out, unsigned short* outb, int D_) {
  const int row = blockIdx.x;
  __shared__ float pre[DD];
  __shared__ float red[256];
  float ls = 0.f;
  for (int d = threadIdx.x; d < D_; d += 256) {
    float v = x[(size_t)row * D_ + d];
    float hv = h[(size_t)row * D_ + d];
    if (mode == 1) hv = eluf(hv);
    v += hv;
    pre[d] = v;
    ls += v;
  }
  red[threadIdx.x] = ls; __syncthreads();
  for (int s = 128; s > 0; s >>= 1) {
    if (threadIdx.x < s) red[threadIdx.x] += red[threadIdx.x + s];
    __syncthreads();
  }
  const float mu = red[0] / D_;
  __syncthreads();
  float lv = 0.f;
  for (int d = threadIdx.x; d < D_; d += 256) { float t = pre[d] - mu; lv += t * t; }
  red[threadIdx.x] = lv; __syncthreads();
  for (int s = 128; s > 0; s >>= 1) {
    if (threadIdx.x < s) red[threadIdx.x] += red[threadIdx.x + s];
    __syncthreads();
  }
  const float inv = rsqrtf(red[0] / D_ + 1e-5f);
  for (int d = threadIdx.x; d < D_; d += 256) {
    float y = (pre[d] - mu) * inv * g[d] + b[d];
    out[(size_t)row * D_ + d] = y;
    if (outb) outb[(size_t)row * D_ + d] = f2bf(y);
  }
}
__global__ void k_count(float* cnt, const int* map, int n) {
  int i = blockIdx.x * 256 + threadIdx.x;
  if (i < n) atomicAdd(&cnt[map[i]], 1.f);
}
__global__ void k_scatter_rows(float* dst, const float* src, const int* map, int D_) {
  int n = blockIdx.x;
  int p = map[n];
  for (int d = threadIdx.x; d < D_; d += 256)
    atomicAdd(&dst[(size_t)p * D_ + d], src[(size_t)n * D_ + d]);
}
__global__ void k_div_rows(float* m, const float* cnt, int rows, int D_) {
  long long i = (long long)blockIdx.x * 256 + threadIdx.x;
  if (i >= (long long)rows * D_) return;
  m[i] /= fmaxf(cnt[i / D_], 1.f);
}
__global__ void k_hu(float* hu1, float* hu2, const float* hmat, const float* a,
                     int rows, int D_) {
  int p = blockIdx.x * 256 + threadIdx.x;
  if (p >= rows) return;
  float s1 = 0.f, s2 = 0.f;
  for (int d = 0; d < D_; ++d) {
    float v = hmat[(size_t)p * D_ + d];
    s1 += v * a[d];
    s2 += v * a[D_ + d];
  }
  hu1[p] = s1; hu2[p] = s2;
}
__global__ void k_doc_edge_e(float* e, const float* hu1, const float* hu2,
                             const int* src, const int* tgt, int ne) {
  int i = blockIdx.x * 256 + threadIdx.x;
  if (i < ne) e[i] = leakyf(hu1[src[i]] + hu2[tgt[i]]);
}
__global__ void k_edge_scatter(float* dst, const float* srcmat, const int* srcIdx,
                               const int* tgtIdx, const float* w, int D_) {
  int e = blockIdx.x;
  float s = w[e];
  if (s == 0.f) return;
  int sr = srcIdx[e], tr = tgtIdx[e];
  for (int d = threadIdx.x; d < D_; d += 256)
    atomicAdd(&dst[(size_t)tr * D_ + d], s * srcmat[(size_t)sr * D_ + d]);
}
__global__ void k_dot_scale(float* sc, const unsigned short* Qb, const unsigned short* Kb,
                            int ne, int D_, float scale) {
  int i = blockIdx.x * 256 + threadIdx.x;
  if (i >= ne) return;
  const unsigned short* q = Qb + (size_t)i * D_;
  const unsigned short* k = Kb + (size_t)i * D_;
  float s = 0.f;
  for (int d = 0; d < D_; ++d) s += bf2f(q[d]) * bf2f(k[d]);
  sc[i] = s * scale;
}
__global__ void k_colmean(float* out, const float* x, int rows, int D_) {
  int d = blockIdx.x * 256 + threadIdx.x;
  if (d >= D_) return;
  float s = 0.f;
  for (int r = 0; r < rows; ++r) s += x[(size_t)r * D_ + d];
  out[d] = s / rows;
}
__global__ void k_gemv(float* out, const float* x, const float* W, const float* bias,
                       int K_, int Nc, int act) {
  int j = blockIdx.x * 256 + threadIdx.x;
  if (j >= Nc) return;
  float s = bias ? bias[j] : 0.f;
  for (int k = 0; k < K_; ++k) s += x[k] * W[(size_t)k * Nc + j];
  if (act == 2) s = fmaxf(s, 0.f);
  out[j] = s;
}

// ---------- host orchestration ----------
extern "C" void kernel_launch(void* const* d_in, const int* in_sizes, int n_in,
                              void* d_out, int out_size, void* d_ws, size_t ws_size,
                              hipStream_t stream) {
  (void)in_sizes; (void)n_in; (void)out_size; (void)ws_size;
  // --- input mapping: top-level dict order; params expands via jax tree-flatten
  //     (dict keys sorted, recursive) at position 2 -> 54 leaves at indices 2..55.
  const float* fx = (const float*)d_in[0];
  const float* ix = (const float*)d_in[1];
  const float* c1b = (const float*)d_in[2];
  const float* c1w = (const float*)d_in[3];
  const float* c2b = (const float*)d_in[4];
  const float* c2w = (const float*)d_in[5];
  // ca_f(6..13) / ca_i(14..21): Wk,Wo,Wq,Wv,bk,bo,bq,bv
  const float* caW[2][4]; const float* cab[2][4];
  for (int v = 0; v < 2; ++v)
    for (int j = 0; j < 4; ++j) {
      caW[v][j] = (const float*)d_in[6 + 8 * v + j];
      cab[v][j] = (const float*)d_in[6 + 8 * v + 4 + j];
    }
  const float* docW = (const float*)d_in[22];
  const float* doca = (const float*)d_in[23];
  const float* ln_doc_b = (const float*)d_in[24]; const float* ln_doc_g = (const float*)d_in[25];
  const float* ln_f1_b  = (const float*)d_in[26]; const float* ln_f1_g  = (const float*)d_in[27];
  const float* ln_f2_b  = (const float*)d_in[28]; const float* ln_f2_g  = (const float*)d_in[29];
  const float* ln_i1_b  = (const float*)d_in[30]; const float* ln_i1_g  = (const float*)d_in[31];
  const float* ln_i2_b  = (const float*)d_in[32]; const float* ln_i2_g  = (const float*)d_in[33];
  const float* pb_ = (const float*)d_in[34];
  const float* pw_ = (const float*)d_in[35];
  // rgat_f(36..45) / rgat_i(46..55): WK,WQ,WR,WV,a_b,a_w,bK,bQ,bR,bV
  struct RG { const float *WK,*WQ,*WR,*WV,*a_b,*a_w,*bK,*bQ,*bR,*bV; } rg[2];
  for (int v = 0; v < 2; ++v) {
    int b0 = 36 + 10 * v;
    rg[v].WK  = (const float*)d_in[b0 + 0]; rg[v].WQ  = (const float*)d_in[b0 + 1];
    rg[v].WR  = (const float*)d_in[b0 + 2]; rg[v].WV  = (const float*)d_in[b0 + 3];
    rg[v].a_b = (const float*)d_in[b0 + 4]; rg[v].a_w = (const float*)d_in[b0 + 5];
    rg[v].bK  = (const float*)d_in[b0 + 6]; rg[v].bQ  = (const float*)d_in[b0 + 7];
    rg[v].bR  = (const float*)d_in[b0 + 8]; rg[v].bV  = (const float*)d_in[b0 + 9];
  }
  const int* fedge = (const int*)d_in[56]; const int* ftype = (const int*)d_in[57];
  const int* iedge = (const int*)d_in[58]; const int* itype = (const int*)d_in[59];
  const int* cfi   = (const int*)d_in[60]; const int* cif   = (const int*)d_in[61];
  const int* pmf   = (const int*)d_in[62]; const int* pmi   = (const int*)d_in[63];
  const int* dedge = (const int*)d_in[64];

  // --- workspace bump allocator (~270 MB) ---
  char* base = (char*)d_ws; size_t off = 0;
  auto alloc = [&](size_t bytes) -> void* {
    void* p = base + off; off = (off + bytes + 255) & ~(size_t)255; return p;
  };
  unsigned short* xb[2]; xb[0] = (unsigned short*)alloc((size_t)NN * DD * 2);
                         xb[1] = (unsigned short*)alloc((size_t)NN * DD * 2);
  float* l1[2];  l1[0]  = (float*)alloc((size_t)NN * DD * 4); l1[1]  = (float*)alloc((size_t)NN * DD * 4);
  unsigned short* l1b[2]; l1b[0] = (unsigned short*)alloc((size_t)NN * DD * 2);
                          l1b[1] = (unsigned short*)alloc((size_t)NN * DD * 2);
  float* l2[2];  l2[0]  = (float*)alloc((size_t)NN * DD * 4); l2[1]  = (float*)alloc((size_t)NN * DD * 4);
  float* hacc  = (float*)alloc((size_t)NN * DD * 4);
  float* tmpnd = (float*)alloc((size_t)NN * DD * 4);
  unsigned short* relb = (unsigned short*)alloc((size_t)EE * DD * 2);
  unsigned short* Qb = relb;                                   // reuse after RGAT stage
  unsigned short* Kb = relb + (size_t)ECC * DD;
  unsigned short* aggb = xb[0];                                // reuse after RGAT stage
  unsigned short* Wt1 = (unsigned short*)alloc((size_t)DD * 2 * DD * 2); // [768,1536]
  unsigned short* Wt2 = (unsigned short*)alloc((size_t)DD * DD * 2);     // [768,768]
  float* u      = (float*)alloc((DD + 1) * 4);
  float* ebuf   = (float*)alloc((size_t)EE * 4);
  float* attn   = (float*)alloc((size_t)EE * 4);
  unsigned int* segkey = (unsigned int*)alloc((size_t)NN * 4);
  float* segsum = (float*)alloc((size_t)NN * 4);
  float* pf     = (float*)alloc((size_t)PP * DD * 4);
  float* pi_    = (float*)alloc((size_t)PP * DD * 4);
  float* para   = (float*)alloc((size_t)PP * DD * 4);
  float* docsum = (float*)alloc((size_t)PP * DD * 4);
  float* hdocf  = (float*)alloc((size_t)PP * DD * 4);
  float* doc    = (float*)alloc((size_t)PP * DD * 4);
  unsigned short* pfb   = (unsigned short*)alloc((size_t)PP * DD * 2);
  unsigned short* pib   = (unsigned short*)alloc((size_t)PP * DD * 2);
  unsigned short* parab = (unsigned short*)alloc((size_t)PP * DD * 2);
  float* cnts = (float*)alloc(PP * 4);
  float* hu1  = (float*)alloc(PP * 4);
  float* hu2  = (float*)alloc(PP * 4);
  float* feats = (float*)alloc(3 * DD * 4);
  float* hcls  = (float*)alloc(DD * 4);

  // --- launch helpers ---
  auto zero_ = [&](void* p, size_t words) {
    k_zero<<<dim3((unsigned)((words + 255) / 256)), 256, 0, stream>>>((unsigned int*)p, (long long)words);
  };
  auto cvt_ = [&](unsigned short* d, const float* s, size_t n) {
    k_cvt<<<dim3((unsigned)((n + 255) / 256)), 256, 0, stream>>>(d, s, (long long)n);
  };
  auto tr_ = [&](unsigned short* d, const float* s, int K_, int Nc) {
    size_t n = (size_t)K_ * Nc;
    k_transpose_cvt<<<dim3((unsigned)((n + 255) / 256)), 256, 0, stream>>>(d, s, K_, Nc);
  };
  auto gemm_ = [&](const unsigned short* A0, const unsigned short* A1,
                   const int* i0, const int* i1, const unsigned short* Wt,
                   const float* bias, int M, int Nc, int K_, int Ks, int act,
                   float* oF, unsigned short* oB,
                   const float* rs, const int* sc, float* ac) {
    dim3 g(Nc / 32, M / 256);   // wave = 32x32 output block, 8 waves/block
    k_wmma_gemm<<<g, 256, 0, stream>>>(A0, A1, i0, i1, DD, Wt, bias, M, Nc, K_, Ks,
                                       act, oF, oB, rs, sc, ac);
  };
  auto softmax_ = [&](const int* etype, int r, const int* seg, int ne, int nseg) {
    zero_(segkey, nseg); zero_(segsum, nseg);
    dim3 g((ne + 255) / 256);
    k_seg_max<<<g, 256, 0, stream>>>(ebuf, etype, r, seg, segkey, ne);
    k_seg_expsum<<<g, 256, 0, stream>>>(ebuf, etype, r, seg, segkey, segsum, attn, ne);
    k_seg_norm<<<g, 256, 0, stream>>>(attn, seg, segsum, ne);
  };

  // ---- stage 0: bf16 copies of inputs ----
  cvt_(xb[0], fx, (size_t)NN * DD);
  cvt_(xb[1], ix, (size_t)NN * DD);

  // ---- stage 1: RGAT per view ----
  const float* xs[2]   = {fx, ix};
  const int*   edg[2]  = {fedge, iedge};
  const int*   ety[2]  = {ftype, itype};
  const float* lng1[2] = {ln_f1_g, ln_i1_g};
  const float* lnb1[2] = {ln_f1_b, ln_i1_b};
  for (int v = 0; v < 2; ++v) {
    const int* src = edg[v];
    const int* tgt = edg[v] + EE;
    zero_(hacc, (size_t)NN * DD);
    for (int r = 0; r < RR; ++r) {
      tr_(Wt1, rg[v].WR + (size_t)r * 2 * DD * DD, 2 * DD, DD);
      tr_(Wt2, rg[v].WV + (size_t)r * DD * DD, DD, DD);
      k_compute_u<<<dim3((DD + 1 + 255) / 256), 256, 0, stream>>>(
          u, rg[v].WQ + (size_t)r * DD * DD, rg[v].WK + (size_t)r * DD * DD,
          rg[v].bQ + (size_t)r * DD, rg[v].bK + (size_t)r * DD,
          rg[v].a_w, rg[v].a_b, DD);
      // rel = leaky( [x[src], x[tgt]] @ WR[r] + bR[r] )  -> bf16
      gemm_(xb[v], xb[v], src, tgt, Wt1, rg[v].bR + (size_t)r * DD,
            EE, DD, 2 * DD, DD, /*leaky*/1, nullptr, relb, nullptr, nullptr, nullptr);
      // e = leaky( rel . u + const )   (Q/K GEMMs algebraically collapsed)
      k_edge_score<<<dim3(EE / 256), 256, 0, stream>>>(ebuf, relb, u, EE, DD);
      softmax_(ety[v], r, tgt, EE, NN);
      // h[tgt] += attn * (rel @ WV[r] + bV[r])   (fused scatter epilogue)
      gemm_(relb, nullptr, nullptr, nullptr, Wt2, rg[v].bV + (size_t)r * DD,
            EE, DD, DD, DD, 0, nullptr, nullptr, attn, tgt, hacc);
    }
    // f1/i1 = layernorm(x + elu(h)), plus bf16 copy
    k_rowops_ln<<<dim3(NN), 256, 0, stream>>>(xs[v], hacc, 1, lng1[v], lnb1[v],
                                              l1[v], l1b[v], DD);
  }

  // ---- stage 2: cross attention per view ----
  const int*   cedg[2] = {cfi, cif};
  const float* lng2[2] = {ln_f2_g, ln_i2_g};
  const float* lnb2[2] = {ln_f2_b, ln_i2_b};
  const float  scal = 1.0f / sqrtf((float)DD);
  for (int v = 0; v < 2; ++v) {
    const int ov = 1 - v;
    const int* qi = cedg[v];
    const int* ki = cedg[v] + ECC;
    // caW: [0]=Wk [1]=Wo [2]=Wq [3]=Wv ; cab: [0]=bk [1]=bo [2]=bq [3]=bv
    tr_(Wt2, caW[v][2], DD, DD);
    gemm_(l1b[v], nullptr, qi, nullptr, Wt2, cab[v][2], ECC, DD, DD, DD, 0,
          nullptr, Qb, nullptr, nullptr, nullptr);
    tr_(Wt2, caW[v][0], DD, DD);
    gemm_(l1b[ov], nullptr, ki, nullptr, Wt2, cab[v][0], ECC, DD, DD, DD, 0,
          nullptr, Kb, nullptr, nullptr, nullptr);
    k_dot_scale<<<dim3(ECC / 256), 256, 0, stream>>>(ebuf, Qb, Kb, ECC, DD, scal);
    softmax_(nullptr, -1, qi, ECC, NN);
    zero_(hacc, (size_t)NN * DD);
    tr_(Wt2, caW[v][3], DD, DD);
    gemm_(l1b[ov], nullptr, ki, nullptr, Wt2, cab[v][3], ECC, DD, DD, DD, 0,
          nullptr, nullptr, attn, qi, hacc);            // agg[qi] += attn * V
    cvt_(aggb, hacc, (size_t)NN * DD);
    tr_(Wt2, caW[v][1], DD, DD);
    gemm_(aggb, nullptr, nullptr, nullptr, Wt2, cab[v][1], NN, DD, DD, DD, 0,
          tmpnd, nullptr, nullptr, nullptr, nullptr);   // agg @ Wo + bo
    k_rowops_ln<<<dim3(NN), 256, 0, stream>>>(l1[v], tmpnd, 2, lng2[v], lnb2[v],
                                              l2[v], nullptr, DD);
  }

  // ---- stage 3: paragraph pooling + projection ----
  zero_(pf, (size_t)PP * DD); zero_(cnts, PP);
  k_count<<<dim3(NN / 256), 256, 0, stream>>>(cnts, pmf, NN);
  k_scatter_rows<<<dim3(NN), 256, 0, stream>>>(pf, l2[0], pmf, DD);
  k_div_rows<<<dim3((PP * DD) / 256), 256, 0, stream>>>(pf, cnts, PP, DD);
  zero_(pi_, (size_t)PP * DD); zero_(cnts, PP);
  k_count<<<dim3(NN / 256), 256, 0, stream>>>(cnts, pmi, NN);
  k_scatter_rows<<<dim3(NN), 256, 0, stream>>>(pi_, l2[1], pmi, DD);
  k_div_rows<<<dim3((PP * DD) / 256), 256, 0, stream>>>(pi_, cnts, PP, DD);
  cvt_(pfb, pf, (size_t)PP * DD);
  cvt_(pib, pi_, (size_t)PP * DD);
  tr_(Wt1, pw_, 2 * DD, DD);
  gemm_(pfb, pib, nullptr, nullptr, Wt1, pb_, PP, DD, 2 * DD, DD, /*relu*/2,
        para, parab, nullptr, nullptr, nullptr);

  // ---- stage 4: doc GAT ----
  tr_(Wt2, docW, DD, DD);
  gemm_(parab, nullptr, nullptr, nullptr, Wt2, nullptr, PP, DD, DD, DD, 0,
        hdocf, nullptr, nullptr, nullptr, nullptr);
  k_hu<<<dim3((PP + 255) / 256), 256, 0, stream>>>(hu1, hu2, hdocf, doca, PP, DD);
  const int* dsrc = dedge;
  const int* dtgt = dedge + EDD;
  k_doc_edge_e<<<dim3(EDD / 256), 256, 0, stream>>>(ebuf, hu1, hu2, dsrc, dtgt, EDD);
  softmax_(nullptr, -1, dtgt, EDD, PP);
  zero_(docsum, (size_t)PP * DD);
  k_edge_scatter<<<dim3(EDD), 256, 0, stream>>>(docsum, hdocf, dsrc, dtgt, attn, DD);
  k_rowops_ln<<<dim3(PP), 256, 0, stream>>>(para, docsum, 1, ln_doc_g, ln_doc_b,
                                            doc, nullptr, DD);

  // ---- stage 5: readout + classifier ----
  k_colmean<<<dim3((DD + 255) / 256), 256, 0, stream>>>(feats, doc, PP, DD);
  k_colmean<<<dim3((DD + 255) / 256), 256, 0, stream>>>(feats + DD, l2[0], NN, DD);
  k_colmean<<<dim3((DD + 255) / 256), 256, 0, stream>>>(feats + 2 * DD, l2[1], NN, DD);
  k_gemv<<<dim3((DD + 255) / 256), 256, 0, stream>>>(hcls, feats, c1w, c1b, 3 * DD, DD, 2);
  k_gemv<<<dim3(1), 256, 0, stream>>>((float*)d_out, hcls, c2w, c2b, DD, 2, 0);
}
